// GraphAttentionLayer_18734647345745
// MI455X (gfx1250) — matrix operand order repaired and checked by reading, not compiled, not run on hardware
//
#include <hip/hip_runtime.h>
#include <hip/hip_bf16.h>
#include <stdint.h>

// ---------------------------------------------------------------------------
// GAT layer for MI455X (gfx1250, wave32):
//   Ht = H @ W^T per head  -> v_wmma_f32_16x16x32_bf16 (f32 accumulate)
//   edge softmax + SpMM    -> L2-resident gathers (Ht = 102MB < 192MB L2),
//                             heads folded into one atomic scatter pass.
// ---------------------------------------------------------------------------

#define FIN   256
#define FOUT  128
#define HEADS 4
#define ALPHA 0.2f
#define EPS   1e-16f

typedef __attribute__((ext_vector_type(16))) __bf16 v16bf;
typedef __attribute__((ext_vector_type(8)))  float  v8f;

// ---- helpers --------------------------------------------------------------

__device__ __forceinline__ unsigned short f32_to_bf16(float f) {
    unsigned u = __float_as_uint(f);
    unsigned r = u + 0x7FFFu + ((u >> 16) & 1u);   // round-to-nearest-even
    return (unsigned short)(r >> 16);
}

// order-preserving float -> uint mapping so we can use u32 atomic max
__device__ __forceinline__ unsigned f2ord(float f) {
    unsigned u = __float_as_uint(f);
    return (u & 0x80000000u) ? ~u : (u | 0x80000000u);
}
__device__ __forceinline__ float ord2f(unsigned u) {
    return (u & 0x80000000u) ? __uint_as_float(u ^ 0x80000000u)
                             : __uint_as_float(~u);
}

// ---- init: emax=ordered(-inf)=0, denom=0, out = bias ----------------------

__global__ void gat_init(unsigned* __restrict__ emax, float* __restrict__ denom,
                         float* __restrict__ out, const float* __restrict__ bias,
                         int N) {
    int i = blockIdx.x * blockDim.x + threadIdx.x;
    if (i < HEADS * N) { emax[i] = 0u; denom[i] = 0.0f; }
    if (i < N * FOUT)  out[i] = bias[i & (FOUT - 1)];
}

// ---- f32 -> bf16 downconvert ----------------------------------------------

__global__ void cvt_bf16(const float* __restrict__ src,
                         unsigned short* __restrict__ dst, int n) {
    int i = blockIdx.x * blockDim.x + threadIdx.x;
    if (i < n) dst[i] = f32_to_bf16(src[i]);
}

// ---- Ht[h] = H @ W[h]^T via v_wmma_f32_16x16x32_bf16 ----------------------
// wave -> one 16x16 output tile; block = 4 waves = 4 f-tiles.
// grid = (ceil(N/16), 2, HEADS)

__global__ __launch_bounds__(128)
void gat_gemm_wmma(const unsigned short* __restrict__ Hbf,
                   const unsigned short* __restrict__ Wbf,
                   float* __restrict__ Ht, int N) {
    const int lane  = threadIdx.x & 31;
    const int wave  = threadIdx.x >> 5;
    const int mtile = blockIdx.x;
    const int ntile = blockIdx.y * 4 + wave;   // 8 column tiles of 16
    const int head  = blockIdx.z;
    const int hi    = lane >> 4;               // lane half select
    const int lmod  = lane & 15;

    int m = (mtile << 4) + lmod;               // A row this lane supplies
    int mclamp = m < N ? m : N - 1;
    const unsigned short* arow = Hbf + (size_t)mclamp * FIN;
    // B(k,f) = W[f][k]: column f of B is a contiguous row of Wbf
    const unsigned short* brow =
        Wbf + ((size_t)head * FOUT + ((ntile << 4) + lmod)) * FIN;

    v8f c = {0.f, 0.f, 0.f, 0.f, 0.f, 0.f, 0.f, 0.f};

#pragma unroll
    for (int k0 = 0; k0 < FIN; k0 += 32) {
        union { v16bf v; uint4 u[2]; } A, B;
        // A frag: elems 0..7 = K hi*8..hi*8+7 ; elems 8..15 = K 16+hi*8..
        A.u[0] = *(const uint4*)(arow + k0 + hi * 8);
        A.u[1] = *(const uint4*)(arow + k0 + 16 + hi * 8);
        // B frag: elems 0..7 = K hi*8..hi*8+7 ; elems 8..15 = K 16+hi*8..
        B.u[0] = *(const uint4*)(brow + k0 + hi * 8);
        B.u[1] = *(const uint4*)(brow + k0 + 16 + hi * 8);
        c = __builtin_amdgcn_wmma_f32_16x16x32_bf16(
                false, A.v, false, B.v, (short)0, c, false, false);
    }

    // D: VGPR v -> row hi*8+v, col lmod
    float* drow = Ht + ((size_t)head * N + (size_t)(mtile << 4)) * FOUT
                     + (ntile << 4) + lmod;
    const int rbase = hi * 8;
#pragma unroll
    for (int v = 0; v < 8; ++v) {
        int row = (mtile << 4) + rbase + v;
        if (row < N) drow[(size_t)(rbase + v) * FOUT] = c[v];
    }
}

// ---- per-node scores: s_src = Ht.a[:F], s_dst = Ht.a[F:] (wave per (n,h)) -

__global__ __launch_bounds__(256)
void gat_scores(const float* __restrict__ Ht, const float* __restrict__ a,
                float* __restrict__ s_src, float* __restrict__ s_dst, int N) {
    int gw   = (blockIdx.x * blockDim.x + threadIdx.x) >> 5;
    int lane = threadIdx.x & 31;
    if (gw >= N * HEADS) return;
    int head = gw & (HEADS - 1);
    int n    = gw >> 2;
    float4 hv = *(const float4*)(Ht + ((size_t)head * N + n) * FOUT + lane * 4);
    float4 a1 = *(const float4*)(a + head * 2 * FOUT + lane * 4);
    float4 a2 = *(const float4*)(a + head * 2 * FOUT + FOUT + lane * 4);
    float d1 = hv.x * a1.x + hv.y * a1.y + hv.z * a1.z + hv.w * a1.w;
    float d2 = hv.x * a2.x + hv.y * a2.y + hv.z * a2.z + hv.w * a2.w;
#pragma unroll
    for (int off = 16; off; off >>= 1) {
        d1 += __shfl_xor(d1, off, 32);
        d2 += __shfl_xor(d2, off, 32);
    }
    if (lane == 0) {
        s_src[head * N + n] = d1;
        s_dst[head * N + n] = d2;
    }
}

// ---- pass 1: segment max over src (thread per edge, all heads) ------------

__global__ void gat_edge_max(const int* __restrict__ src, const int* __restrict__ dst,
                             const float* __restrict__ s_src, const float* __restrict__ s_dst,
                             unsigned* __restrict__ emax, int N, int E) {
    int i = blockIdx.x * blockDim.x + threadIdx.x;
    if (i >= E) return;
    int s = src[i], d = dst[i];
#pragma unroll
    for (int h = 0; h < HEADS; ++h) {
        float e = s_src[h * N + s] + s_dst[h * N + d];
        e = e > 0.0f ? e : ALPHA * e;
        atomicMax(emax + h * N + s, f2ord(e));
    }
}

// ---- pass 2: e_exp + segment sum ------------------------------------------

__global__ void gat_edge_expsum(const int* __restrict__ src, const int* __restrict__ dst,
                                const float* __restrict__ s_src, const float* __restrict__ s_dst,
                                const unsigned* __restrict__ emax,
                                float* __restrict__ denom, float* __restrict__ eexp,
                                int N, int E) {
    int i = blockIdx.x * blockDim.x + threadIdx.x;
    if (i >= E) return;
    int s = src[i], d = dst[i];
    float4 o;
#pragma unroll
    for (int h = 0; h < HEADS; ++h) {
        float e = s_src[h * N + s] + s_dst[h * N + d];
        e = e > 0.0f ? e : ALPHA * e;
        float ex = __expf(e - ord2f(emax[h * N + s]));
        (&o.x)[h] = ex;
        atomicAdd(denom + h * N + s, ex);
    }
    *(float4*)(eexp + (size_t)i * HEADS) = o;
}

// ---- pass 3: out[src] += sum_h attn_h * Ht[h][dst] / HEADS (wave per edge)-

__global__ __launch_bounds__(256)
void gat_edge_scatter(const int* __restrict__ src, const int* __restrict__ dst,
                      const float* __restrict__ eexp, const float* __restrict__ denom,
                      const float* __restrict__ Ht, float* __restrict__ out,
                      int N, int E) {
    int gw   = (blockIdx.x * blockDim.x + threadIdx.x) >> 5;
    int lane = threadIdx.x & 31;
    if (gw >= E) return;
    int s = src[gw], d = dst[gw];
    float4 ex  = *(const float4*)(eexp + (size_t)gw * HEADS);
    float4 acc = {0.f, 0.f, 0.f, 0.f};
    const float* hbase = Ht + (size_t)d * FOUT + lane * 4;   // L2-resident
#pragma unroll
    for (int h = 0; h < HEADS; ++h) {
        float attn = (&ex.x)[h] / (denom[h * N + s] + EPS) * (1.0f / HEADS);
        float4 hv = *(const float4*)(hbase + (size_t)h * N * FOUT);
        acc.x += attn * hv.x;  acc.y += attn * hv.y;
        acc.z += attn * hv.z;  acc.w += attn * hv.w;
    }
    float* op = out + (size_t)s * FOUT + lane * 4;
    atomicAdd(op + 0, acc.x);
    atomicAdd(op + 1, acc.y);
    atomicAdd(op + 2, acc.z);
    atomicAdd(op + 3, acc.w);
}

// ---------------------------------------------------------------------------

extern "C" void kernel_launch(void* const* d_in, const int* in_sizes, int n_in,
                              void* d_out, int out_size, void* d_ws, size_t ws_size,
                              hipStream_t stream) {
    const float* H    = (const float*)d_in[0];
    const int*   ei   = (const int*)  d_in[1];
    const float* W    = (const float*)d_in[2];
    const float* a    = (const float*)d_in[3];
    const float* bias = (const float*)d_in[4];
    float* out = (float*)d_out;

    const int N = in_sizes[0] / FIN;
    const int E = in_sizes[1] / 2;
    const int* src = ei;
    const int* dst = ei + E;

    // workspace carve-out (256B aligned)
    char* ws = (char*)d_ws;
    size_t off = 0;
    auto carve = [&](size_t bytes) -> void* {
        void* p = ws + off;
        off = (off + bytes + 255) & ~(size_t)255;
        return p;
    };
    unsigned short* Hbf  = (unsigned short*)carve((size_t)N * FIN * 2);
    unsigned short* Wbf  = (unsigned short*)carve((size_t)HEADS * FOUT * FIN * 2);
    float*          Ht   = (float*)carve((size_t)HEADS * N * FOUT * 4);
    float*          ssrc = (float*)carve((size_t)HEADS * N * 4);
    float*          sdst = (float*)carve((size_t)HEADS * N * 4);
    unsigned*       emax = (unsigned*)carve((size_t)HEADS * N * 4);
    float*          den  = (float*)carve((size_t)HEADS * N * 4);
    float*          eexp = (float*)carve((size_t)E * HEADS * 4);
    (void)ws_size; (void)n_in; (void)out_size;

    const int T = 256;

    // init accumulators + out=bias
    {
        int n = N * FOUT;
        gat_init<<<(n + T - 1) / T, T, 0, stream>>>(emax, den, out, bias, N);
    }
    // downconvert H, W to bf16
    {
        int n = N * FIN;
        cvt_bf16<<<(n + T - 1) / T, T, 0, stream>>>(H, Hbf, n);
        int nw = HEADS * FOUT * FIN;
        cvt_bf16<<<(nw + T - 1) / T, T, 0, stream>>>(W, Wbf, nw);
    }
    // WMMA GEMM: Ht[h] = H @ W[h]^T
    {
        dim3 grid((N + 15) / 16, 2, HEADS);
        gat_gemm_wmma<<<grid, 128, 0, stream>>>(Hbf, Wbf, Ht, N);
    }
    // attention scores
    {
        int waves = N * HEADS;
        gat_scores<<<(waves * 32 + T - 1) / T, T, 0, stream>>>(Ht, a, ssrc, sdst, N);
    }
    // edge passes
    gat_edge_max<<<(E + T - 1) / T, T, 0, stream>>>(src, dst, ssrc, sdst, emax, N, E);
    gat_edge_expsum<<<(E + T - 1) / T, T, 0, stream>>>(src, dst, ssrc, sdst, emax, den, eexp, N, E);
    {
        long long threads = (long long)E * 32;
        gat_edge_scatter<<<(int)((threads + T - 1) / T), T, 0, stream>>>(
            src, dst, eexp, den, Ht, out, N, E);
    }
}
